// SynthesisLayer_3152505995679
// MI455X (gfx1250) — matrix operand order repaired and checked by reading, not compiled, not run on hardware
//
#include <hip/hip_runtime.h>
#include <hip/hip_bf16.h>

typedef __attribute__((ext_vector_type(16))) __bf16 v16bf;
typedef __attribute__((ext_vector_type(8)))  __bf16 v8bf;
typedef __attribute__((ext_vector_type(8)))  float  v8f;

#define NB    8
#define CIN   512
#define COUT  512
#define WDIM  512
#define HW    4096
#define LDP   40   // LDS pitch in bf16 elems: 80B rows -> 16B-aligned b128 frag loads

__device__ __forceinline__ unsigned short f2bf(float f) {
  unsigned u = __builtin_bit_cast(unsigned, f);
  u += 0x7FFFu + ((u >> 16) & 1u);           // round-to-nearest-even
  return (unsigned short)(u >> 16);
}

// Async DMA: 16B per lane, global -> LDS, no staging VGPRs (tracked by ASYNCcnt)
__device__ __forceinline__ void async_copy_b128(const unsigned short* gptr,
                                                unsigned short* lptr) {
  auto lds = (__attribute__((address_space(3))) unsigned short*)lptr;
  asm volatile("global_load_async_to_lds_b128 %0, %1, off"
               :: "v"(lds), "v"(gptr) : "memory");
}

__device__ __forceinline__ void wait_async_all() {
  asm volatile("s_wait_asynccnt 0" ::: "memory");
}

// ---------------- styles = w_latent @ (affine_w^T / sqrt(512)) + affine_b ----
__global__ void styles_kernel(const float* __restrict__ wl,
                              const float* __restrict__ aw,
                              const float* __restrict__ ab,
                              float* __restrict__ styles) {
  int idx = blockIdx.x * 256 + threadIdx.x;       // 0..4095
  int b = idx >> 9, i = idx & 511;
  const float4* wv = (const float4*)(wl + (size_t)b * WDIM);
  const float4* av = (const float4*)(aw + (size_t)i * WDIM);
  float s = 0.f;
#pragma unroll 8
  for (int j = 0; j < WDIM / 4; ++j) {
    float4 a = av[j], w = wv[j];
    s += a.x * w.x + a.y * w.y + a.z * w.z + a.w * w.w;
  }
  styles[idx] = s * 0.044194173824159216f + ab[i];  // 1/sqrt(512)
}

// ---------------- wmod[b][o][:] = demod(weight[o]*styles[b]) as bf16 --------
__global__ void wmod_kernel(const float* __restrict__ weight,
                            const float* __restrict__ styles,
                            unsigned short* __restrict__ wmodb) {
  int gw = blockIdx.x * 8 + (threadIdx.x >> 5);   // one wave per (b,o)
  int lane = threadIdx.x & 31;
  int b = gw >> 9, o = gw & 511;
  const float4* wr = (const float4*)(weight + (size_t)o * CIN) + lane * 4;
  const float4* sr = (const float4*)(styles + (size_t)b * CIN) + lane * 4;
  float v[16];
  float ss = 0.f;
#pragma unroll
  for (int j = 0; j < 4; ++j) {
    float4 w = wr[j], s = sr[j];
    v[j * 4 + 0] = w.x * s.x; v[j * 4 + 1] = w.y * s.y;
    v[j * 4 + 2] = w.z * s.z; v[j * 4 + 3] = w.w * s.w;
    ss += v[j*4+0]*v[j*4+0] + v[j*4+1]*v[j*4+1] + v[j*4+2]*v[j*4+2] + v[j*4+3]*v[j*4+3];
  }
#pragma unroll
  for (int off = 16; off > 0; off >>= 1) ss += __shfl_xor(ss, off, 32);
  float d = __frsqrt_rn(ss + 1e-8f);
  unsigned pk[8];
#pragma unroll
  for (int j = 0; j < 8; ++j)
    pk[j] = (unsigned)f2bf(v[2*j] * d) | ((unsigned)f2bf(v[2*j+1] * d) << 16);
  uint4* dst = (uint4*)(wmodb + (size_t)gw * CIN + lane * 16);
  dst[0] = make_uint4(pk[0], pk[1], pk[2], pk[3]);
  dst[1] = make_uint4(pk[4], pk[5], pk[6], pk[7]);
}

// ---------------- XT[b][hw][cin] (bf16) = transpose(x[b][cin][hw]) ----------
__global__ void xpose_kernel(const float* __restrict__ x,
                             unsigned short* __restrict__ xt) {
  __shared__ float t[32][33];
  int b = blockIdx.z, kt = blockIdx.y, nt = blockIdx.x;
  int col = threadIdx.x & 31, rg = threadIdx.x >> 5;
  const float* src = x + ((size_t)b * CIN + kt * 32) * HW + nt * 32;
#pragma unroll
  for (int j = 0; j < 4; ++j) {
    int r = rg + 8 * j;
    t[r][col] = src[(size_t)r * HW + col];
  }
  __syncthreads();
  unsigned short* dst = xt + ((size_t)b * HW + nt * 32) * CIN + kt * 32;
#pragma unroll
  for (int j = 0; j < 4; ++j) {
    int n = rg + 8 * j;
    dst[(size_t)n * CIN + col] = f2bf(t[col][n]);   // stride-33 LDS read: conflict-free
  }
}

// ---------------- Y[b] = lrelu(Wmod[b] @ X[b] + bias) * sqrt(2) -------------
// Block tile 128(M) x 256(N), 8 waves in 2x4 grid, 64x64 per wave:
// 16 v_wmma per K-slab against 16 ds_load_b128 -> 2x math density vs 64x32.
__global__ __launch_bounds__(256)
void gemm_kernel(const unsigned short* __restrict__ wmodb,
                 const unsigned short* __restrict__ xt,
                 const float* __restrict__ bias,
                 float* __restrict__ out) {
  __shared__ unsigned short ldsA[2][128 * LDP];   // 20 KB
  __shared__ unsigned short ldsB[2][256 * LDP];   // 40 KB

  const int tid = threadIdx.x;
  const int b  = blockIdx.z;
  const int m0 = blockIdx.y * 128;   // COUT tile
  const int n0 = blockIdx.x * 256;   // pixel tile
  const int wave = tid >> 5, lane = tid & 31;
  const int wm = wave >> 2, wn = wave & 3;        // 2x4 wave grid, 64x64 per wave
  const int half = lane >> 4, lrow = lane & 15;

  const unsigned short* gA = wmodb + ((size_t)b * COUT + m0) * CIN;
  const unsigned short* gB = xt    + ((size_t)b * HW   + n0) * CIN;

  // per-thread copy slices: A = 128 rows x 64B (2 x b128), B = 256 rows x 64B (4 x b128)
  auto issue_tile = [&](int k0, int buf) {
#pragma unroll
    for (int p = 0; p < 2; ++p) {
      int e = tid + p * 256, r = e >> 2, sg = e & 3;
      async_copy_b128(gA + (size_t)r * CIN + k0 + sg * 8, &ldsA[buf][r * LDP + sg * 8]);
    }
#pragma unroll
    for (int p = 0; p < 4; ++p) {
      int e = tid + p * 256, r = e >> 2, sg = e & 3;
      async_copy_b128(gB + (size_t)r * CIN + k0 + sg * 8, &ldsB[buf][r * LDP + sg * 8]);
    }
  };

  v8f acc[4][4] = {};

  issue_tile(0, 0);
  wait_async_all();
  __syncthreads();

  for (int kk = 0; kk < 16; ++kk) {
    const int cur = kk & 1;
    if (kk < 15) issue_tile((kk + 1) * 32, cur ^ 1);  // DMA next slab during math

    v16bf afrag[4], bfrag[4];
#pragma unroll
    for (int tm = 0; tm < 4; ++tm) {
      // A-frag: lanes 0-15 M=lrow K{0..7,16..23}; lanes 16-31 K{8..15,24..31}
      int base = (wm * 64 + tm * 16 + lrow) * LDP + 8 * half;
      v8bf lo = *(const v8bf*)&ldsA[cur][base];
      v8bf hi = *(const v8bf*)&ldsA[cur][base + 16];
      afrag[tm] = __builtin_shufflevector(lo, hi, 0,1,2,3,4,5,6,7,8,9,10,11,12,13,14,15);
    }
#pragma unroll
    for (int tn = 0; tn < 4; ++tn) {
      // B-frag: lane n=lrow, lanes 0-15 K{0..7,16..23}; lanes 16-31 K{8..15,24..31}
      int base = (wn * 64 + tn * 16 + lrow) * LDP + 8 * half;
      v8bf lo = *(const v8bf*)&ldsB[cur][base];
      v8bf hi = *(const v8bf*)&ldsB[cur][base + 16];
      bfrag[tn] = __builtin_shufflevector(lo, hi, 0,1,2,3,4,5,6,7,8,9,10,11,12,13,14,15);
    }
#pragma unroll
    for (int tm = 0; tm < 4; ++tm)
#pragma unroll
      for (int tn = 0; tn < 4; ++tn)
        acc[tm][tn] = __builtin_amdgcn_wmma_f32_16x16x32_bf16(
            false, afrag[tm], false, bfrag[tn], (short)0, acc[tm][tn], false, false);

    if (kk < 15) {
      wait_async_all();     // our slab for kk+1 is in LDS
      __syncthreads();      // everyone's slab is in LDS
    }
  }

  // fused epilogue: bias + lrelu(0.2) * sqrt(2)
#pragma unroll
  for (int tm = 0; tm < 4; ++tm) {
#pragma unroll
    for (int tn = 0; tn < 4; ++tn) {
#pragma unroll
      for (int r = 0; r < 8; ++r) {
        int o = m0 + wm * 64 + tm * 16 + r + 8 * half;  // C/D: M = r + 8*half
        int p = n0 + wn * 64 + tn * 16 + lrow;          //      N = lane & 15
        float v = acc[tm][tn][r] + bias[o];
        v = (v >= 0.f ? v : 0.2f * v) * 1.4142135623730951f;
        out[((size_t)b * COUT + o) * HW + p] = v;
      }
    }
  }
}

extern "C" void kernel_launch(void* const* d_in, const int* in_sizes, int n_in,
                              void* d_out, int out_size, void* d_ws, size_t ws_size,
                              hipStream_t stream) {
  const float* x  = (const float*)d_in[0];
  const float* wl = (const float*)d_in[1];
  const float* aw = (const float*)d_in[2];
  const float* ab = (const float*)d_in[3];
  const float* wt = (const float*)d_in[4];
  const float* bs = (const float*)d_in[5];
  float* out = (float*)d_out;

  float* styles         = (float*)d_ws;
  unsigned short* wmodb = (unsigned short*)((char*)d_ws + (1 << 16));
  unsigned short* xtb   = (unsigned short*)((char*)d_ws + (1 << 16)
                                            + (size_t)NB * COUT * CIN * 2);

  styles_kernel<<<16, 256, 0, stream>>>(wl, aw, ab, styles);
  wmod_kernel<<<512, 256, 0, stream>>>(wt, styles, wmodb);
  xpose_kernel<<<dim3(HW / 32, CIN / 32, NB), 256, 0, stream>>>(x, xtb);
  gemm_kernel<<<dim3(HW / 256, COUT / 128, NB), 256, 0, stream>>>(wmodb, xtb, bs, out);
}
